// MoE_32658931319292
// MI455X (gfx1250) — compile-verified, matrix operand order checked by reading, not run on hardware
//
#include <hip/hip_runtime.h>

// ---------------------------------------------------------------------------
// MoE top-2 (T=4096, D=1024, H=2048, O=1024, E=8) for MI455X / gfx1250.
// bf16 WMMA (v_wmma_f32_16x16x32_bf16) with f32 accumulation.
// A tiles staged via gfx1250 async global->LDS copies (ASYNCcnt),
// B tiles staged transposed+converted through VGPRs.
// x is pre-converted to bf16 once so GEMM inner loops carry no conversions.
// ws layout (assumes ws_size >= ~43 MB):
//   [0,32K)        int2   gate_idx[T]
//   [32K,64K)      float2 gate_val[T]
//   [64K,192K)     int    tokenlist[E*T]
//   [192K,320K)    float  gatelist[E*T]
//   [320K,..)      int    counts[E]
//   [512K,8.5M)    bf16   xbf[T*D]       (8 MB)
//   [8.5M,42M)     bf16   hbuf[T*2*H]    (~32 MB)
// ---------------------------------------------------------------------------

typedef __attribute__((ext_vector_type(16))) __bf16 v16bf;
typedef __attribute__((ext_vector_type(8)))  float  v8f;

#define T_TOK 4096
#define D_IN  1024
#define H_MID 2048
#define O_OUT 1024
#define E_NUM 8

#define TM 64      // token tile
#define TN 128     // output-feature tile
#define KB 32      // k-step (one WMMA K)

#define WS_GIDX   0u
#define WS_GVAL   32768u
#define WS_TLIST  65536u
#define WS_GLIST  196608u
#define WS_CNT    327680u
#define WS_XBF    524288u
#define WS_HBUF   (524288u + 8388608u)

union FragBF { v16bf v; uint4 q[2]; };

__device__ __forceinline__ unsigned pack2(float a, float b) {
  union { __bf16 h[2]; unsigned u; } z;
  z.h[0] = (__bf16)a; z.h[1] = (__bf16)b;
  return z.u;
}

// gfx1250 async global->LDS 128-bit copy (tracked by ASYNCcnt)
__device__ __forceinline__ void async_copy_b128(unsigned lds_off, const void* gaddr) {
  asm volatile("global_load_async_to_lds_b128 %0, %1, off"
               :: "v"(lds_off), "v"(gaddr) : "memory");
}
__device__ __forceinline__ void wait_async0() {
  asm volatile("s_wait_asynccnt 0x0" ::: "memory");
}
__device__ __forceinline__ unsigned lds_off_of(const void* p) {
  return (unsigned)(uintptr_t)p;   // LDS aperture: low 32 bits are the LDS offset
}

// ---------------------------------------------------------------- zero output
__global__ void moe_zero(float4* __restrict__ p, int n4) {
  int i = blockIdx.x * blockDim.x + threadIdx.x;
  if (i < n4) p[i] = float4{0.f, 0.f, 0.f, 0.f};
}

// ------------------------------------------------------- x -> bf16 streaming
__global__ void moe_cvt_x(const float4* __restrict__ x, uint4* __restrict__ xbf,
                          int n8) {
  const int i = blockIdx.x * blockDim.x + threadIdx.x;   // 8 floats per thread
  if (i >= n8) return;
  const float4 f0 = x[i * 2 + 0];
  const float4 f1 = x[i * 2 + 1];
  uint4 u;
  u.x = pack2(f0.x, f0.y); u.y = pack2(f0.z, f0.w);
  u.z = pack2(f1.x, f1.y); u.w = pack2(f1.z, f1.w);
  xbf[i] = u;
}

// -------------------------------------------------------------------- gating
// one wave32 per token: logits = x @ Wg + bg ; top-2 ; softmax
__global__ __launch_bounds__(256) void moe_gate(
    const float* __restrict__ x, const float* __restrict__ Wg,
    const float* __restrict__ bg, int2* __restrict__ gidx,
    float2* __restrict__ gval) {
  const int wave = threadIdx.x >> 5;
  const int lane = threadIdx.x & 31;
  const int t = blockIdx.x * 8 + wave;
  const float* xr = x + (size_t)t * D_IN;
  float acc[E_NUM] = {};
  for (int i = 0; i < D_IN / 32; ++i) {
    const int d = i * 32 + lane;
    const float xv = xr[d];
    const float4 w0 = *(const float4*)(Wg + (size_t)d * E_NUM);
    const float4 w1 = *(const float4*)(Wg + (size_t)d * E_NUM + 4);
    acc[0] += xv * w0.x; acc[1] += xv * w0.y;
    acc[2] += xv * w0.z; acc[3] += xv * w0.w;
    acc[4] += xv * w1.x; acc[5] += xv * w1.y;
    acc[6] += xv * w1.z; acc[7] += xv * w1.w;
  }
#pragma unroll
  for (int off = 16; off > 0; off >>= 1)
#pragma unroll
    for (int e = 0; e < E_NUM; ++e)
      acc[e] += __shfl_down(acc[e], off, 32);
  if (lane == 0) {
    float v0 = -1e30f, v1 = -1e30f; int i0 = 0, i1 = 0;
#pragma unroll
    for (int e = 0; e < E_NUM; ++e) {
      const float v = acc[e] + bg[e];
      if (v > v0)      { v1 = v0; i1 = i0; v0 = v; i0 = e; }
      else if (v > v1) { v1 = v;  i1 = e; }
    }
    const float ex = __expf(v1 - v0);
    const float g0 = 1.f / (1.f + ex);
    gidx[t] = make_int2(i0, i1);
    gval[t] = make_float2(g0, 1.f - g0);
  }
}

// ------------------------------------------------------------------- routing
// one wave per expert: ballot-compact token list (deterministic order)
__global__ void moe_route(const int2* __restrict__ gidx,
                          const float2* __restrict__ gval,
                          int* __restrict__ tlist, float* __restrict__ glist,
                          int* __restrict__ counts) {
  const int e = blockIdx.x;
  const int lane = threadIdx.x;
  int cnt = 0;
  for (int base = 0; base < T_TOK; base += 32) {
    const int t = base + lane;
    const int2 gi = gidx[t];
    const float2 gv = gval[t];
    bool m = false; float g = 0.f;
    if (gi.x == e)      { m = true; g = gv.x; }
    else if (gi.y == e) { m = true; g = gv.y; }
    const unsigned mask = (unsigned)__ballot(m);
    if (m) {
      const int pos = cnt + __popc(mask & ((1u << lane) - 1u));
      tlist[e * T_TOK + pos] = t;
      glist[e * T_TOK + pos] = g;
    }
    cnt += __popc(mask);
  }
  if (lane == 0) counts[e] = cnt;
}

// shared tile: B^T staged as packed bf16 pairs (uint4 view -> ds_load_b128),
// then reused as f32 C restage buffer after the K loop.
struct SMTile {
  union {
    uint4    b4[TN][5];        // [n][k], pitch 80 B (16B aligned)
    unsigned b[TN][20];        // scalar store view, same layout
    float    c[TM][TN + 4];    // padded pitch to spread banks
  };
};

// -------------------------------------------------------- GEMM1: h = relu(xW1+b1)
__global__ __launch_bounds__(256) void moe_gemm1(
    const __bf16* __restrict__ xbf, const float* __restrict__ W1,
    const float* __restrict__ b1, const int* __restrict__ tlist,
    const int2* __restrict__ gidx, const int* __restrict__ counts,
    __bf16* __restrict__ hbuf) {
  const int e = blockIdx.z;
  const int cnt = counts[e];
  const int m0 = blockIdx.y * TM;
  if (m0 >= cnt) return;
  const int nb = blockIdx.x * TN;

  __shared__ SMTile sm;
  __shared__ uint4 smA[TM][5];   // bf16 x tile, pitch 80 B (bank-spread)

  const int tid = threadIdx.x;
  const int lane = tid & 31;
  const int wv = tid >> 5;
  const int wm = wv & 1;     // 2 wave rows  (32 tokens each)
  const int wn = wv >> 1;    // 4 wave cols  (32 features each)
  const int l15 = lane & 15;
  const int hi = lane >> 4;

  // async A staging role: one 16B chunk per thread covers 64x32 bf16
  const int am = tid >> 2;           // 0..63 staged token row
  const int ach = tid & 3;           // chunk 0..3
  const int aslot = m0 + am;
  const int atok = (aslot < cnt) ? tlist[e * T_TOK + aslot] : 0;
  const char* agrow = (const char*)(xbf + (size_t)atok * D_IN);
  const unsigned alds = lds_off_of(&smA[am][ach]);

  const float* w_e = W1 + (size_t)e * D_IN * H_MID;
  const int nstage = tid & (TN - 1);
  const int half = tid >> 7;                 // 0..1 -> k half
  const int nglob = nb + nstage;

  v8f c[2][2] = {};

  for (int kb = 0; kb < D_IN; kb += KB) {
    __syncthreads();
    // async-stage A tile (64 x 32 bf16): straight copy, fragment-ready
    async_copy_b128(alds, agrow + (size_t)kb * 2 + ach * 16);
    // stage W1 k-slice transposed into LDS: sm.b[n][k] (bf16 pairs)
#pragma unroll
    for (int kk = 0; kk < 8; ++kk) {
      const int k = half * 16 + kk * 2;
      const float* wp = w_e + (size_t)(kb + k) * H_MID + nglob;
      sm.b[nstage][half * 8 + kk] = pack2(wp[0], wp[H_MID]);
    }
    if (kb + KB < D_IN)
      __builtin_prefetch(w_e + (size_t)(kb + KB + half * 16) * H_MID + nglob, 0, 1);
    wait_async0();
    __syncthreads();

    // A fragments straight from LDS (bf16, row-major, b128 loads)
    FragBF a[2];
#pragma unroll
    for (int i = 0; i < 2; ++i) {
      const int m = wm * 32 + i * 16 + l15;
      a[i].q[0] = smA[m][hi];
      a[i].q[1] = smA[m][2 + hi];
    }
    // B fragments from LDS (column n contiguous over K) as b128 loads
    FragBF bf[2];
#pragma unroll
    for (int j = 0; j < 2; ++j) {
      const int n = wn * 32 + j * 16 + l15;
      bf[j].q[0] = sm.b4[n][hi];
      bf[j].q[1] = sm.b4[n][2 + hi];
    }
#pragma unroll
    for (int i = 0; i < 2; ++i)
#pragma unroll
      for (int j = 0; j < 2; ++j)
        c[i][j] = __builtin_amdgcn_wmma_f32_16x16x32_bf16(
            false, a[i].v, false, bf[j].v, (short)0, c[i][j], false, false);
  }

  // restage C through LDS for coalesced h writes
  __syncthreads();
#pragma unroll
  for (int i = 0; i < 2; ++i)
#pragma unroll
    for (int j = 0; j < 2; ++j)
#pragma unroll
      for (int v = 0; v < 8; ++v)
        sm.c[wm * 32 + i * 16 + hi * 8 + v][wn * 32 + j * 16 + l15] = c[i][j][v];
  __syncthreads();

  const int row = tid >> 2;
  const int cg = (tid & 3) * 32;
  const int slot = m0 + row;
  if (slot < cnt) {
    const int tok = tlist[e * T_TOK + slot];
    const int2 gi = gidx[tok];
    const int s = (gi.x == e) ? 0 : 1;
    __bf16* hp = hbuf + ((size_t)tok * 2 + s) * H_MID + nb + cg;
    const float* bp = b1 + (size_t)e * H_MID + nb + cg;
#pragma unroll
    for (int c0 = 0; c0 < 32; c0 += 8) {
      uint4 u;
      u.x = pack2(fmaxf(sm.c[row][cg + c0 + 0] + bp[c0 + 0], 0.f),
                  fmaxf(sm.c[row][cg + c0 + 1] + bp[c0 + 1], 0.f));
      u.y = pack2(fmaxf(sm.c[row][cg + c0 + 2] + bp[c0 + 2], 0.f),
                  fmaxf(sm.c[row][cg + c0 + 3] + bp[c0 + 3], 0.f));
      u.z = pack2(fmaxf(sm.c[row][cg + c0 + 4] + bp[c0 + 4], 0.f),
                  fmaxf(sm.c[row][cg + c0 + 5] + bp[c0 + 5], 0.f));
      u.w = pack2(fmaxf(sm.c[row][cg + c0 + 6] + bp[c0 + 6], 0.f),
                  fmaxf(sm.c[row][cg + c0 + 7] + bp[c0 + 7], 0.f));
      *(uint4*)(hp + c0) = u;
    }
  }
}

// ---------------------------------------------- GEMM2: out += gate*(hW2+b2)
__global__ __launch_bounds__(256) void moe_gemm2(
    const __bf16* __restrict__ hbuf, const float* __restrict__ W2,
    const float* __restrict__ b2, const int* __restrict__ tlist,
    const float* __restrict__ glist, const int2* __restrict__ gidx,
    const int* __restrict__ counts, float* __restrict__ out) {
  const int e = blockIdx.z;
  const int cnt = counts[e];
  const int m0 = blockIdx.y * TM;
  if (m0 >= cnt) return;
  const int nb = blockIdx.x * TN;

  __shared__ SMTile sm;
  __shared__ uint4 smA[TM][5];   // bf16 h tile, pitch 80 B (bank-spread)

  const int tid = threadIdx.x;
  const int lane = tid & 31;
  const int wv = tid >> 5;
  const int wm = wv & 1;
  const int wn = wv >> 1;
  const int l15 = lane & 15;
  const int hi = lane >> 4;

  // async A staging role: one 16B chunk per thread covers 64x32 bf16
  const int am = tid >> 2;
  const int ach = tid & 3;
  const int aslot = m0 + am;
  const int atok = (aslot < cnt) ? tlist[e * T_TOK + aslot] : 0;
  int as = 0;
  if (aslot < cnt) { const int2 gi = gidx[atok]; as = (gi.x == e) ? 0 : 1; }
  const char* agrow = (const char*)(hbuf + ((size_t)atok * 2 + as) * H_MID);
  const unsigned alds = lds_off_of(&smA[am][ach]);

  const float* w_e = W2 + (size_t)e * H_MID * O_OUT;
  const int nstage = tid & (TN - 1);
  const int half = tid >> 7;
  const int nglob = nb + nstage;

  v8f c[2][2] = {};

  for (int kb = 0; kb < H_MID; kb += KB) {
    __syncthreads();
    // async-stage A tile (64 x 32 bf16): straight copy, fragment-ready
    async_copy_b128(alds, agrow + (size_t)kb * 2 + ach * 16);
    // stage W2 k-slice transposed into LDS
#pragma unroll
    for (int kk = 0; kk < 8; ++kk) {
      const int k = half * 16 + kk * 2;
      const float* wp = w_e + (size_t)(kb + k) * O_OUT + nglob;
      sm.b[nstage][half * 8 + kk] = pack2(wp[0], wp[O_OUT]);
    }
    if (kb + KB < H_MID)
      __builtin_prefetch(w_e + (size_t)(kb + KB + half * 16) * O_OUT + nglob, 0, 1);
    wait_async0();
    __syncthreads();

    // A fragments straight from LDS (bf16, row-major, b128 loads)
    FragBF a[2];
#pragma unroll
    for (int i = 0; i < 2; ++i) {
      const int m = wm * 32 + i * 16 + l15;
      a[i].q[0] = smA[m][hi];
      a[i].q[1] = smA[m][2 + hi];
    }
    FragBF bf[2];
#pragma unroll
    for (int j = 0; j < 2; ++j) {
      const int n = wn * 32 + j * 16 + l15;
      bf[j].q[0] = sm.b4[n][hi];
      bf[j].q[1] = sm.b4[n][2 + hi];
    }
#pragma unroll
    for (int i = 0; i < 2; ++i)
#pragma unroll
      for (int j = 0; j < 2; ++j)
        c[i][j] = __builtin_amdgcn_wmma_f32_16x16x32_bf16(
            false, a[i].v, false, bf[j].v, (short)0, c[i][j], false, false);
  }

  __syncthreads();
#pragma unroll
  for (int i = 0; i < 2; ++i)
#pragma unroll
    for (int j = 0; j < 2; ++j)
#pragma unroll
      for (int v = 0; v < 8; ++v)
        sm.c[wm * 32 + i * 16 + hi * 8 + v][wn * 32 + j * 16 + l15] = c[i][j][v];
  __syncthreads();

  const int row = tid >> 2;
  const int cg = (tid & 3) * 32;
  const int slot = m0 + row;
  if (slot < cnt) {
    const int tok = tlist[e * T_TOK + slot];
    const float gate = glist[e * T_TOK + slot];
    float* op = out + (size_t)tok * O_OUT + nb + cg;
    const float* bp = b2 + (size_t)e * O_OUT + nb + cg;
#pragma unroll
    for (int c0 = 0; c0 < 32; ++c0)
      atomicAdd(op + c0, (sm.c[row][cg + c0] + bp[c0]) * gate);
  }
}

// ------------------------------------------------------------------- launch
extern "C" void kernel_launch(void* const* d_in, const int* in_sizes, int n_in,
                              void* d_out, int out_size, void* d_ws, size_t ws_size,
                              hipStream_t stream) {
  const float* x  = (const float*)d_in[0];
  const float* Wg = (const float*)d_in[1];
  const float* bg = (const float*)d_in[2];
  const float* W1 = (const float*)d_in[3];
  const float* b1 = (const float*)d_in[4];
  const float* W2 = (const float*)d_in[5];
  const float* b2 = (const float*)d_in[6];
  float* out = (float*)d_out;

  char* ws = (char*)d_ws;
  int2*   gidx  = (int2*)(ws + WS_GIDX);
  float2* gval  = (float2*)(ws + WS_GVAL);
  int*    tlist = (int*)(ws + WS_TLIST);
  float*  glist = (float*)(ws + WS_GLIST);
  int*    counts = (int*)(ws + WS_CNT);
  __bf16* xbf   = (__bf16*)(ws + WS_XBF);
  __bf16* hbuf  = (__bf16*)(ws + WS_HBUF);

  moe_zero<<<(T_TOK * O_OUT / 4 + 255) / 256, 256, 0, stream>>>(
      (float4*)out, T_TOK * O_OUT / 4);
  moe_cvt_x<<<(T_TOK * D_IN / 8 + 255) / 256, 256, 0, stream>>>(
      (const float4*)x, (uint4*)xbf, T_TOK * D_IN / 8);
  moe_gate<<<T_TOK / 8, 256, 0, stream>>>(x, Wg, bg, gidx, gval);
  moe_route<<<E_NUM, 32, 0, stream>>>(gidx, gval, tlist, glist, counts);
  moe_gemm1<<<dim3(H_MID / TN, T_TOK / TM, E_NUM), 256, 0, stream>>>(
      xbf, W1, b1, tlist, gidx, counts, hbuf);
  moe_gemm2<<<dim3(O_OUT / TN, T_TOK / TM, E_NUM), 256, 0, stream>>>(
      hbuf, W2, b2, tlist, glist, gidx, counts, out);
}
